// MatTrans_36112085025172
// MI455X (gfx1250) — matrix operand order repaired and checked by reading, not compiled, not run on hardware
//
#include <hip/hip_runtime.h>
#include <math.h>

// ---------------- problem constants (match reference) ----------------
#define MULC      128
#define DIMC      3200
#define NUM_MULC  640
#define NN        4096
#define NE        16384
#define EDGE_DIM  20
#define WNUMEL    5376   // 42 paths * MUL

__device__ __constant__ int   c_OFF[5]   = {0, 128, 512, 1152, 2048};
// alpha[lo] = sqrt((2lo+1)/paths_per_lo[lo]); ppl = {5,8,10,10,9}
__device__ __constant__ float c_ALPHA[5] = {0.4472135955f, 0.6123724357f,
                                            0.7071067812f, 0.8366600265f, 1.0f};

typedef __attribute__((ext_vector_type(16))) _Float16     v16h;
typedef __attribute__((ext_vector_type(8)))  _Float16     v8h;
typedef __attribute__((ext_vector_type(8)))  float        v8f;
typedef __attribute__((ext_vector_type(4)))  unsigned int v4u;
typedef __attribute__((ext_vector_type(4)))  int          v4i;
typedef __attribute__((ext_vector_type(8)))  int          v8i_t;

__device__ inline float silu_f(float x) {
  // x * sigmoid(x) with fast rcp (v_rcp_f32) instead of IEEE divide
  return x * __builtin_amdgcn_rcpf(1.0f + __expf(-x));
}

// =====================================================================
// TDM: DMA one 16-row x 32-col f32 tile (row stride = strideElems) from
// global into LDS at ldsAddr, padded +4 dwords per 32-dword row so the
// LDS tile is [16][36] f32 (144B row pitch: 16B-aligned, conflict-free).
// D# fields per CDNA5 ISA 8.3/8.4: group0 {count=1, lds_addr, global
// tile addr, type=2}; group1 {data_size=4B, pad_enable, pad_interval=32dw
// (code 4), pad_amount=4dw (code 3), tensor=32x16, tile=32x16, stride}.
// =====================================================================
__device__ inline void tdm_load_tile_f32(unsigned ldsAddr, const float* gptr,
                                         int strideElems)
{
  unsigned long long ga = (unsigned long long)(size_t)gptr;
  v4u g0;
  g0[0] = 1u;                                   // count=1 valid descriptor
  g0[1] = ldsAddr;                              // LDS byte address
  g0[2] = (unsigned)ga;                         // global_addr[31:0]
  g0[3] = ((unsigned)(ga >> 32) & 0x01FFFFFFu)  // global_addr[56:32]
          | 0x80000000u;                        // type=2 ("image")
  v8i_t g1;
  g1[0] = (int)((2u << 16)      // data_size = 4B
              | (1u << 20)      // pad_enable
              | (4u << 22)      // pad_interval code 4 = 32 dwords
              | (3u << 25));    // pad_amount  code 3 = 4 dwords
  g1[1] = (int)(32u << 16);     // tensor_dim0 = 32 (low16 in [31:16])
  g1[2] = (int)(16u << 16);     // tensor_dim0 hi=0 | tensor_dim1 = 16
  g1[3] = (int)(32u << 16);     // tensor_dim1 hi=0 | tile_dim0 = 32
  g1[4] = 16;                   // tile_dim1 = 16 | tile_dim2 = 0
  g1[5] = strideElems;          // tensor_dim0_stride[31:0]
  g1[6] = 0;                    // stride hi | tensor_dim1_stride lo
  g1[7] = 0;
  v4i g2 = {0, 0, 0, 0};
  v4i g3 = {0, 0, 0, 0};
#if defined(__clang_major__) && (__clang_major__ >= 23)
  v8i_t g4 = {0, 0, 0, 0, 0, 0, 0, 0};
  __builtin_amdgcn_tensor_load_to_lds(g0, g1, g2, g3, g4, 0);
#else
  __builtin_amdgcn_tensor_load_to_lds(g0, g1, g2, g3, 0);
#endif
}

// =====================================================================
// Generic GEMM  C[M,N] = act(A[M,K] @ B[K,N] + bias)
//   act: 0 = none, 1 = silu, 2 = multiply into existing C (C *= v)
// f32 in/out, f16 WMMA (f32 accum).  Block = 16 rows x 128 cols (8 waves,
// one 16x16 tile each).  A tile DMA'd by the Tensor Data Mover into LDS
// (double-buffered, wave 0 issues, TENSORcnt synchronized); B fragment
// streamed from global (L2-resident weights).  Caller guarantees
// K % 32 == 0, M % 16 == 0, (N/16) % 8 == 0.
// =====================================================================
__global__ void gemm_kernel(const float* __restrict__ A, const float* __restrict__ B,
                            const float* __restrict__ bias, float* __restrict__ C,
                            int M, int N, int K, int act)
{
  __shared__ __align__(16) float As[2][16][36];    // TDM-padded tiles
  const int tid    = threadIdx.x;
  const int waveId = tid >> 5;
  const int lane   = tid & 31;
  const int m16    = lane & 15;
  const int half   = lane >> 4;
  const int tm     = blockIdx.y;
  const int tn     = blockIdx.x * 8 + waveId;
  const int colB   = tn * 16 + m16;

  const float* aTile = A + (size_t)tm * 16 * K;    // + k0 per step

  if (waveId == 0)
    tdm_load_tile_f32((unsigned)(size_t)&As[0][0][0], aTile, K);

  v8f acc = {};
  for (int k0 = 0; k0 < K; k0 += 32) {
    const int cur = (k0 >> 5) & 1;
    if (waveId == 0) __builtin_amdgcn_s_wait_tensorcnt(0);
    __syncthreads();
    if (k0 + 32 < K && waveId == 0)
      tdm_load_tile_f32((unsigned)(size_t)&As[cur ^ 1][0][0], aTile + k0 + 32, K);

    // A fragment: 4x 16B LDS chunks (ISA 16-bit A 16x32 layout), cvt to f16
    const float* arow = &As[cur][m16][0];
    float4 q0 = *(const float4*)(arow + half * 8);
    float4 q1 = *(const float4*)(arow + half * 8 + 4);
    float4 q2 = *(const float4*)(arow + 16 + half * 8);
    float4 q3 = *(const float4*)(arow + 16 + half * 8 + 4);
    v16h a, b;
    a[0]  = (_Float16)q0.x; a[1]  = (_Float16)q0.y;
    a[2]  = (_Float16)q0.z; a[3]  = (_Float16)q0.w;
    a[4]  = (_Float16)q1.x; a[5]  = (_Float16)q1.y;
    a[6]  = (_Float16)q1.z; a[7]  = (_Float16)q1.w;
    a[8]  = (_Float16)q2.x; a[9]  = (_Float16)q2.y;
    a[10] = (_Float16)q2.z; a[11] = (_Float16)q2.w;
    a[12] = (_Float16)q3.x; a[13] = (_Float16)q3.y;
    a[14] = (_Float16)q3.z; a[15] = (_Float16)q3.w;

    // B fragment: branch-free strided global loads (batched under one wait)
    const float* bp = B + (size_t)(k0 + half * 8) * N + colB;
#pragma unroll
    for (int e = 0; e < 16; ++e) b[e] = (_Float16)bp[(size_t)(e + (e >> 3) * 8) * N];
    __builtin_prefetch(bp + (size_t)32 * N, 0, 1);   // global_prefetch_b8

    acc = __builtin_amdgcn_wmma_f32_16x16x32_f16(false, a, false, b, (short)0,
                                                 acc, false, false);
  }

#pragma unroll
  for (int r = 0; r < 8; ++r) {
    int row = tm * 16 + r + half * 8;                // f32 C/D layout
    size_t oi = (size_t)row * N + colB;
    float v = acc[r] + bias[colB];
    if (act == 1) v = silu_f(v);
    if (act == 2) C[oi] *= v; else C[oi] = v;
  }
}

// =====================================================================
// o3_linear (one l per launch):  y[n,v,m] = inv * sum_u W[u,v] x[n,u,m]
//   (+ b0[v] for l==0, + resid).  Rows = (n,m) pairs, K = 128.
// Block = 16 rows x 128 cols; strided (dim-gather) A-tile shared via LDS,
// double-buffered, K-loop fully unrolled (4 WMMAs).  TDM not applicable
// (strided transpose-gather), so threads stage cooperatively.
// =====================================================================
__global__ void o3_linear_kernel(const float* __restrict__ x, const float* __restrict__ W,
                                 const float* __restrict__ b0, const float* __restrict__ resid,
                                 float* __restrict__ out, int count, int off, int dim)
{
  __shared__ _Float16 As[2][16][40];
  const float inv = 0.0883883476483184f;           // 1/sqrt(128)
  const int tid  = threadIdx.x;
  const int lane = tid & 31;
  const int m16  = lane & 15;
  const int half = lane >> 4;
  const int tm   = blockIdx.x;
  const int tn   = tid >> 5;                       // 8 waves cover all 128 cols
  const int colB = tn * 16 + m16;

  const int sr = tid >> 4;
  const int sc = (tid & 15) * 2;
  const int rG = tm * 16 + sr;
  const int nS = rG / dim, mS = rG % dim;
  const float* base = x + (size_t)nS * DIMC + off + mS;   // element u at stride dim

  auto stage = [&](int buf, int k0) {
    As[buf][sr][sc]     = (_Float16)base[(k0 + sc) * dim];
    As[buf][sr][sc + 1] = (_Float16)base[(k0 + sc + 1) * dim];
  };

  stage(0, 0);
  v8f acc = {};
#pragma unroll
  for (int kk = 0; kk < 4; ++kk) {
    const int k0 = kk * 32, cur = kk & 1;
    __syncthreads();
    if (kk < 3) stage(cur ^ 1, k0 + 32);

    v8h a0 = *(const v8h*)&As[cur][m16][half * 8];
    v8h a1 = *(const v8h*)&As[cur][m16][16 + half * 8];
    v16h a, b;
#pragma unroll
    for (int e = 0; e < 8; ++e) { a[e] = a0[e]; a[e + 8] = a1[e]; }

    const float* bp = W + (size_t)(k0 + half * 8) * 128 + colB;
#pragma unroll
    for (int e = 0; e < 16; ++e) b[e] = (_Float16)bp[(e + (e >> 3) * 8) * 128];

    acc = __builtin_amdgcn_wmma_f32_16x16x32_f16(false, a, false, b, (short)0,
                                                 acc, false, false);
  }

#pragma unroll
  for (int r = 0; r < 8; ++r) {
    int row = tm * 16 + r + half * 8;
    int nO = row / dim, mO = row % dim;
    size_t oi = (size_t)nO * DIMC + off + colB * dim + mO;
    float v = acc[r] * inv;
    if (b0)    v += b0[colB];
    if (resid) v += resid[oi];
    out[oi] = v;
  }
}

// =====================================================================
// NormGate features: f0[n, l*128+u] = scalar (l=0) or irrep norm (l>=1)
// =====================================================================
__global__ void f0_kernel(const float* __restrict__ x, float* __restrict__ f0, int count)
{
  int tid = blockIdx.x * blockDim.x + threadIdx.x;
  if (tid >= count * NUM_MULC) return;
  int n = tid / NUM_MULC, c = tid % NUM_MULC;
  int l = c >> 7, u = c & 127;
  const float* xr = x + (size_t)n * DIMC;
  if (l == 0) { f0[tid] = xr[u]; return; }
  int dim = 2 * l + 1, off = c_OFF[l];
  float s = 0.0f;
  for (int m = 0; m < dim; ++m) { float v = xr[off + u * dim + m]; s += v * v; }
  f0[tid] = sqrtf(s + 1e-12f);
}

// gates: scalars replaced by g[:, :128]; l>=1 irreps scaled by g[:, l*128+u]
__global__ void gate_apply_kernel(const float* __restrict__ x, const float* __restrict__ g,
                                  float* __restrict__ out, int count)
{
  int tid = blockIdx.x * blockDim.x + threadIdx.x;
  if (tid >= count * DIMC) return;
  int n = tid / DIMC, c = tid % DIMC;
  int l, off;
  if      (c < 128)  { l = 0; off = 0; }
  else if (c < 512)  { l = 1; off = 128; }
  else if (c < 1152) { l = 2; off = 512; }
  else if (c < 2048) { l = 3; off = 1152; }
  else               { l = 4; off = 2048; }
  if (l == 0) { out[tid] = g[(size_t)n * NUM_MULC + c]; return; }
  int dim = 2 * l + 1;
  int u = (c - off) / dim;
  out[tid] = x[tid] * g[(size_t)n * NUM_MULC + l * 128 + u];
}

// s0[e] = [xi[:128], xj[:128], per-irrep dots(l>=1)]  -> [E, 768]
__global__ void s0_dots_kernel(const float* __restrict__ nf, const int* __restrict__ src,
                               const int* __restrict__ dst, float* __restrict__ s0)
{
  int tid = blockIdx.x * blockDim.x + threadIdx.x;
  if (tid >= NE * 6 * MULC) return;
  int e = tid / (6 * MULC), j = tid % (6 * MULC);
  const float* xi = nf + (size_t)src[e] * DIMC;
  const float* xj = nf + (size_t)dst[e] * DIMC;
  float v;
  if (j < 128)      v = xi[j];
  else if (j < 256) v = xj[j - 128];
  else {
    int c = j - 256, li = c / 128 + 1, u = c % 128;
    int dim = 2 * li + 1, off = c_OFF[li];
    float s = 0.0f;
    for (int m = 0; m < dim; ++m)
      s += xi[off + u * dim + m] * xj[off + u * dim + m];
    v = s;
  }
  s0[tid] = v;
}

// pad A columns: out[M,K2] <- in[M,K1], zero-filled
__global__ void pad_cols_kernel(const float* __restrict__ in, float* __restrict__ out,
                                int M, int K1, int K2)
{
  int tid = blockIdx.x * blockDim.x + threadIdx.x;
  if (tid >= M * K2) return;
  int r = tid / K2, c = tid % K2;
  out[tid] = (c < K1) ? in[(size_t)r * K1 + c] : 0.0f;
}

// pad B rows: out[K2,N] <- in[K1,N], zero-filled
__global__ void pad_rows_kernel(const float* __restrict__ in, float* __restrict__ out,
                                int K1, int K2, int N)
{
  int tid = blockIdx.x * blockDim.x + threadIdx.x;
  if (tid >= K2 * N) return;
  int k = tid / N, n = tid % N;
  out[tid] = (k < K1) ? in[(size_t)k * N + n] : 0.0f;
}

// =====================================================================
// On-device real Wigner-3j generation (one-time deterministic init).
// =====================================================================
__device__ double factd(int n) { double r = 1.0; for (int i = 2; i <= n; ++i) r *= i; return r; }

__device__ double cgf(int j1, int m1, int j2, int m2, int j3, int m3)
{
  if (m1 + m2 != m3) return 0.0;
  int lo = j1 - j2; if (lo < 0) lo = -lo;
  if (j3 < lo || j3 > j1 + j2) return 0.0;
  double pre = sqrt((double)(2 * j3 + 1) * factd(j1 + j2 - j3) * factd(j1 - j2 + j3) *
                    factd(-j1 + j2 + j3) / factd(j1 + j2 + j3 + 1));
  pre *= sqrt(factd(j1 + m1) * factd(j1 - m1) * factd(j2 + m2) * factd(j2 - m2) *
              factd(j3 + m3) * factd(j3 - m3));
  double s = 0.0;
  for (int k = 0; k <= j1 + j2 - j3; ++k) {
    int d2 = j1 + j2 - j3 - k, d3 = j1 - m1 - k, d4 = j2 + m2 - k,
        d5 = j3 - j2 + m1 + k, d6 = j3 - j1 - m2 + k;
    if (d2 < 0 || d3 < 0 || d4 < 0 || d5 < 0 || d6 < 0) continue;
    double den = factd(k) * factd(d2) * factd(d3) * factd(d4) * factd(d5) * factd(d6);
    s += ((k & 1) ? -1.0 : 1.0) / den;
  }
  return pre * s;
}

__device__ void fillU(int l, double (*re)[9], double (*im)[9])
{
  for (int a = 0; a < 9; ++a)
    for (int b = 0; b < 9; ++b) { re[a][b] = 0.0; im[a][b] = 0.0; }
  re[l][l] = 1.0;
  const double s2 = 0.7071067811865476;
  for (int m = 1; m <= l; ++m) {
    double sgn = (m & 1) ? -1.0 : 1.0;
    re[l + m][l + m] = sgn * s2;
    re[l + m][l - m] = s2;
    im[l - m][l - m] = s2;
    im[l - m][l + m] = -sgn * s2;
  }
}

__global__ void w3j_init_kernel(float* __restrict__ w3j)
{
  int p = threadIdx.x;
  int pidx = 0, off = 0, L1 = -1, L2 = 0, L3 = 0, myoff = 0;
  for (int l1 = 0; l1 <= 4; ++l1)
    for (int l2 = 0; l2 <= 4; ++l2) {
      int lmin = (l1 > l2) ? l1 - l2 : l2 - l1;
      int lmax = (l1 + l2 > 4) ? 4 : l1 + l2;
      for (int lo = lmin; lo <= lmax; ++lo) {
        if ((l1 + l2 + lo) & 1) continue;
        if (pidx == p) { L1 = l1; L2 = l2; L3 = lo; myoff = off; }
        off += (2 * l1 + 1) * (2 * l2 + 1) * (2 * lo + 1);
        ++pidx;
      }
    }
  if (L1 < 0) return;

  int d1 = 2 * L1 + 1, d2 = 2 * L2 + 1, d3 = 2 * L3 + 1;
  double U1re[9][9], U1im[9][9], U2re[9][9], U2im[9][9], U3re[9][9], U3im[9][9];
  fillU(L1, U1re, U1im); fillU(L2, U2re, U2im); fillU(L3, U3re, U3im);

  double Ct[9][9];
  for (int i = 0; i < d1; ++i)
    for (int j = 0; j < d2; ++j) {
      int k = i - L1 + j - L2 + L3;
      Ct[i][j] = (k >= 0 && k < d3) ? cgf(L1, i - L1, L2, j - L2, L3, k - L3) : 0.0;
    }

  float Rre[729], Rim[729];
  double sre = 0.0, sim = 0.0;
  for (int a = 0; a < d1; ++a)
    for (int b = 0; b < d2; ++b)
      for (int c = 0; c < d3; ++c) {
        double are = 0.0, aim = 0.0;
        for (int i = 0; i < d1; ++i) {
          double u1re = U1re[a][i], u1im = -U1im[a][i];   // conj
          for (int j = 0; j < d2; ++j) {
            double cv = Ct[i][j];
            if (cv == 0.0) continue;
            int k = i - L1 + j - L2 + L3;
            double u2re = U2re[b][j], u2im = -U2im[b][j]; // conj
            double u3re = U3re[c][k], u3im = U3im[c][k];
            double t1re = u1re * u2re - u1im * u2im;
            double t1im = u1re * u2im + u1im * u2re;
            double t2re = t1re * u3re - t1im * u3im;
            double t2im = t1re * u3im + t1im * u3re;
            are += t2re * cv; aim += t2im * cv;
          }
        }
        int o = (a * d2 + b) * d3 + c;
        Rre[o] = (float)are; Rim[o] = (float)aim;
        sre += fabs(are); sim += fabs(aim);
      }
  int tot = d1 * d2 * d3;
  bool useRe = (sre >= sim);
  double nrm = 0.0;
  for (int o = 0; o < tot; ++o) { double v = useRe ? Rre[o] : Rim[o]; nrm += v * v; }
  nrm = sqrt(nrm) + 1e-30;
  float* dstp = w3j + myoff;
  for (int o = 0; o < tot; ++o)
    dstp[o] = (float)((useRe ? (double)Rre[o] : (double)Rim[o]) / nrm);
}

// =====================================================================
// CG 'uuu' tensor product.  One thread per (n,u).
// Dynamically-indexed per-thread arrays live in LDS (lane-striped
// [idx][256] layout: the dynamic index is wave-uniform -> conflict-free),
// exploiting the 320KB/WGP LDS instead of scratch.  w3j addresses are
// wave-uniform -> scalar loads.
// =====================================================================
#define TP_LDS_FLOATS (3 * 45 * 256)
__global__ void tp_kernel(const float* __restrict__ x1, const float* __restrict__ x2,
                          const int* __restrict__ idx1, const int* __restrict__ idx2,
                          const float* __restrict__ w, int w_stride,
                          const float* __restrict__ w3j, const float* __restrict__ resid,
                          float* __restrict__ out, int count)
{
  extern __shared__ float smem[];
  const int tl = threadIdx.x;                 // lane-striped column
  float* sc1 = smem;                          // [45][256]
  float* sc2 = smem + 45 * 256;               // [45][256]
  float* so  = smem + 2 * 45 * 256;           // [45][256]

  int tid = blockIdx.x * blockDim.x + tl;
  if (tid >= count * MULC) return;
  int n = tid / MULC, u = tid % MULC;
  int r1 = idx1 ? idx1[n] : n;
  int r2 = idx2 ? idx2[n] : n;

  for (int i = 0; i < 45; ++i) so[i * 256 + tl] = 0.0f;
  for (int l = 0; l < 5; ++l) {
    int dim = 2 * l + 1, off = c_OFF[l];
    const float* p1 = x1 + (size_t)r1 * DIMC + off + u * dim;
    const float* p2 = x2 + (size_t)r2 * DIMC + off + u * dim;
    for (int m = 0; m < dim; ++m) {
      sc1[(l * 9 + m) * 256 + tl] = p1[m];
      sc2[(l * 9 + m) * 256 + tl] = p2[m];
    }
  }

  int pidx = 0, w3off = 0;
  for (int l1 = 0; l1 <= 4; ++l1)
    for (int l2 = 0; l2 <= 4; ++l2) {
      int lmin = (l1 > l2) ? l1 - l2 : l2 - l1;
      int lmax = (l1 + l2 > 4) ? 4 : l1 + l2;
      for (int lo = lmin; lo <= lmax; ++lo) {
        if ((l1 + l2 + lo) & 1) continue;
        int d1 = 2 * l1 + 1, d2 = 2 * l2 + 1, d3 = 2 * lo + 1;
        float wv = (w_stride == 0) ? w[pidx * MULC + u]
                                   : w[(size_t)n * w_stride + pidx * MULC + u];
        float coef = c_ALPHA[lo] * wv;
        const float* w3 = w3j + w3off;
        float* orow = so + lo * 9 * 256 + tl;
        for (int i = 0; i < d1; ++i) {
          float a = coef * sc1[(l1 * 9 + i) * 256 + tl];
          for (int j = 0; j < d2; ++j) {
            float prod = a * sc2[(l2 * 9 + j) * 256 + tl];
            const float* w3r = w3 + (i * d2 + j) * d3;
            for (int m = 0; m < d3; ++m)
              orow[m * 256] = fmaf(prod, w3r[m], orow[m * 256]);
          }
        }
        w3off += d1 * d2 * d3;
        ++pidx;
      }
    }

  for (int lo = 0; lo < 5; ++lo) {
    int dim = 2 * lo + 1, off = c_OFF[lo];
    float* op = out + (size_t)n * DIMC + off + u * dim;
    const float* rp = resid ? resid + (size_t)n * DIMC + off + u * dim : nullptr;
    for (int m = 0; m < dim; ++m)
      op[m] = so[(lo * 9 + m) * 256 + tl] + (rp ? rp[m] : 0.0f);
  }
}

// =====================================================================
// Host orchestration
// =====================================================================
extern "C" void kernel_launch(void* const* d_in, const int* in_sizes, int n_in,
                              void* d_out, int out_size, void* d_ws, size_t ws_size,
                              hipStream_t stream)
{
  (void)in_sizes; (void)n_in; (void)out_size; (void)ws_size;
  auto in = [&](int i) { return (const float*)d_in[i]; };

  // ---- inputs: setup_inputs() insertion order, params depth-first ----
  const float* node_feat = in(0);
  const float* edge_attr = in(1);
  const int*   edge_index = (const int*)d_in[2];
  const float* fii = in(3);
  const float* fij = in(4);
  const float *sL_w1 = in(5),  *sL_b1 = in(6),  *sL_w2 = in(7),  *sL_b2 = in(8);
  const float *sR_w1 = in(9),  *sR_b1 = in(10), *sR_w2 = in(11), *sR_b2 = in(12);
  const float *sP_w1 = in(13), *sP_b1 = in(14), *sP_w2 = in(15), *sP_b2 = in(16);
  const float *lin_l_w = in(17), *lin_l_b = in(18);
  const float *lin_r_w = in(19), *lin_r_b = in(20);
  const float *lin_p_w = in(21), *lin_p_b = in(22);
  const float *tp_w_shared = in(23);
  const float *pPre_w1 = in(24), *pPre_b1 = in(25), *pPre_w2 = in(26), *pPre_b2 = in(27);
  const float *pPo_w1  = in(28), *pPo_b1  = in(29), *pPo_w2  = in(30), *pPo_b2  = in(31);
  const float *lin_pre_w  = in(32), *lin_pre_b  = in(33);
  const float *lin_post_w = in(34), *lin_post_b = in(35);
  const float *es_w1 = in(36), *es_b1 = in(37), *es_w2 = in(38), *es_b2 = in(39);
  const float *er_w1 = in(40), *er_b1 = in(41), *er_w2 = in(42), *er_b2 = in(43);

  const int* srcI = edge_index;
  const int* dstI = edge_index + NE;

  float* out_fii = (float*)d_out;
  float* out_fij = out_fii + (size_t)NN * DIMC;

  // ---- workspace carve-up ----
  size_t wo = 0;
  auto alloc = [&](size_t nf) {
    float* p = (float*)((char*)d_ws + wo);
    wo += nf * sizeof(float);
    wo = (wo + 255) & ~(size_t)255;
    return p;
  };
  float* W3   = alloc(32768);
  float* F0   = alloc((size_t)NE * NUM_MULC);
  float* Hb   = alloc((size_t)NE * NUM_MULC);
  float* Gb   = alloc((size_t)NE * NUM_MULC);
  float* GN   = alloc((size_t)NN * DIMC);
  float* XL   = alloc((size_t)NN * DIMC);
  float* XR   = alloc((size_t)NN * DIMC);
  float* Tb   = alloc((size_t)NN * DIMC);
  float* XP   = alloc((size_t)NN * DIMC);
  float* S0   = alloc((size_t)NE * 6 * MULC);
  float* TPW  = alloc((size_t)NE * WNUMEL);
  float* Fb   = alloc((size_t)NE * DIMC);
  float* EA32 = alloc((size_t)NE * 32);
  float* ERW1 = alloc((size_t)32 * 128);

  static const int h_OFF[5] = {0, 128, 512, 1152, 2048};

  auto gemm = [&](const float* A, const float* B, const float* bias, float* C,
                  int M, int N, int K, int act) {
    dim3 grid((N / 16) / 8, M / 16);
    gemm_kernel<<<grid, 256, 0, stream>>>(A, B, bias, C, M, N, K, act);
  };
  auto o3 = [&](const float* xin, const float* Wbase, const float* b0,
                const float* resid, float* outp, int count) {
    for (int l = 0; l < 5; ++l) {
      int dim = 2 * l + 1;
      o3_linear_kernel<<<count * dim / 16, 256, 0, stream>>>(
          xin, Wbase + l * MULC * MULC, (l == 0) ? b0 : nullptr,
          resid, outp, count, h_OFF[l], dim);
    }
  };
  auto norm_gate = [&](const float* xin, int count,
                       const float* w1, const float* b1,
                       const float* w2, const float* b2, float* gated) {
    int t1 = count * NUM_MULC;
    f0_kernel<<<(t1 + 255) / 256, 256, 0, stream>>>(xin, F0, count);
    gemm(F0, w1, b1, Hb, count, NUM_MULC, NUM_MULC, 1);
    gemm(Hb, w2, b2, Gb, count, NUM_MULC, NUM_MULC, 0);
    int t2 = count * DIMC;
    gate_apply_kernel<<<(t2 + 255) / 256, 256, 0, stream>>>(xin, Gb, gated, count);
  };
  auto tp = [&](const float* x1, const float* x2, const int* i1, const int* i2,
                const float* w, int wstride, const float* resid, float* outp, int count) {
    int total = count * MULC;
    tp_kernel<<<(total + 255) / 256, 256, TP_LDS_FLOATS * sizeof(float), stream>>>(
        x1, x2, i1, i2, w, wstride, W3, resid, outp, count);
  };

  // ---- one-time deterministic init ----
  w3j_init_kernel<<<1, 64, 0, stream>>>(W3);

  // ================= SelfLayer =================
  norm_gate(node_feat, NN, sL_w1, sL_b1, sL_w2, sL_b2, GN);
  o3(GN, lin_l_w, lin_l_b, nullptr, XL, NN);
  norm_gate(node_feat, NN, sR_w1, sR_b1, sR_w2, sR_b2, GN);
  o3(GN, lin_r_w, lin_r_b, nullptr, XR, NN);
  tp(XL, XR, nullptr, nullptr, tp_w_shared, 0, node_feat, Tb, NN);  // + residual
  norm_gate(Tb, NN, sP_w1, sP_b1, sP_w2, sP_b2, GN);
  o3(GN, lin_p_w, lin_p_b, fii, out_fii, NN);                       // fii_new

  // ================= PairLayer =================
  {
    int t = NE * 6 * MULC;
    s0_dots_kernel<<<(t + 255) / 256, 256, 0, stream>>>(node_feat, srcI, dstI, S0);
  }
  gemm(S0, es_w1, es_b1, Hb, NE, 128, 6 * MULC, 1);        // silu(s0@es_w1+b1)
  gemm(Hb, es_w2, es_b2, TPW, NE, WNUMEL, 128, 0);         // -> es out

  {
    int t1 = NE * 32;
    pad_cols_kernel<<<(t1 + 255) / 256, 256, 0, stream>>>(edge_attr, EA32, NE, EDGE_DIM, 32);
    int t2 = 32 * 128;
    pad_rows_kernel<<<(t2 + 255) / 256, 256, 0, stream>>>(er_w1, ERW1, EDGE_DIM, 32, 128);
  }
  gemm(EA32, ERW1, er_b1, Hb, NE, 128, 32, 1);             // silu(ea@er_w1+b1), padded K
  gemm(Hb, er_w2, er_b2, TPW, NE, WNUMEL, 128, 2);         // TPW *= er out

  norm_gate(node_feat, NN, pPre_w1, pPre_b1, pPre_w2, pPre_b2, GN);
  o3(GN, lin_pre_w, lin_pre_b, nullptr, XP, NN);
  tp(XP, XP, srcI, dstI, TPW, WNUMEL, nullptr, Fb, NE);    // gathered TP per edge
  norm_gate(Fb, NE, pPo_w1, pPo_b1, pPo_w2, pPo_b2, Fb);   // in-place gate
  o3(Fb, lin_post_w, lin_post_b, fij, out_fij, NE);        // fij_new
}
